// WavefrontEngine_35433480192846
// MI455X (gfx1250) — compile-verified, matrix operand order checked by reading, not run on hardware
//
#include <hip/hip_runtime.h>
#include <hip/hip_bf16.h>

// WavefrontEngine on gfx1250 (MI455X):
//  - 63 stream-ordered tick launches realize the anti-diagonal dependency.
//  - One 256-thread block (8 wave32) per cell; bf16 WMMA 16x16x32, fp32 accum.
//  - Weights read from HBM exactly once (~452 MB -> ~19.4 us floor @ 23.3 TB/s).
//  - Fragments loaded as ds_load_b128/b64 (ISA layout halves are contiguous),
//    B-fragments double-buffered so LDS latency overlaps the WMMAs.

#define GDIM 32
#define DDIM 192
#define HDIM 192
#define BDIM 128

typedef __bf16 bf16_t;
typedef bf16_t v16bf __attribute__((ext_vector_type(16)));
typedef float  v8f   __attribute__((ext_vector_type(8)));
typedef float         fvec2 __attribute__((ext_vector_type(2)));
typedef unsigned int  uivec4 __attribute__((ext_vector_type(4)));
typedef unsigned int  uivec2 __attribute__((ext_vector_type(2)));

union Frag {
  unsigned int u[8];
  uivec4 q[2];
  uivec2 d[4];
  v16bf v;
};

__device__ __forceinline__ unsigned short f2bf(float f) {
  // round-to-nearest-even fp32 -> bf16
  unsigned int u = __float_as_uint(f);
  u += 0x7FFFu + ((u >> 16) & 1u);
  return (unsigned short)(u >> 16);
}

__device__ __forceinline__ unsigned int packbf(float lo, float hi) {
  return (unsigned int)f2bf(lo) | ((unsigned int)f2bf(hi) << 16);
}

__device__ __forceinline__ float gelu_exact(float x) {
  return 0.5f * x * (1.0f + erff(x * 0.70710678118654752f));
}

// LDS layout (bytes), phases separated by barriers:
//   [0, 50176)      phase1: activation chunk 128 x 40 bf16 (10240 B used)
//                   phase2: h buffer        128 x 196 bf16 (50176 B)
//   [50176, 65536)  weight chunk transposed 192 x 40 bf16 (15360 B)
#define HBUF_STRIDE 196   // byte stride 392 (mult of 8 -> b64 frags, bank-clean)
#define CHUNK_STRIDE 40   // byte stride 80 (mult of 16 -> b128 frags, bank-clean)
#define WOFF_BYTES 50176
#define SMEM_BYTES 65536

__device__ __forceinline__ void load_bfrag(Frag& f, const unsigned short* wS,
                                           int nt, int m, int koff) {
  const unsigned short* bb = wS + (nt * 16 + m) * CHUNK_STRIDE + koff;
  f.q[0] = *(const uivec4*)(bb);
  f.q[1] = *(const uivec4*)(bb + 16);
}

__global__ __launch_bounds__(256) void wavefront_cell_kernel(
    const float* __restrict__ x, const float* __restrict__ Wl,
    const float* __restrict__ Wu, const float* __restrict__ bvec,
    const float* __restrict__ V, float* __restrict__ out,
    int tick, int i0) {
  __shared__ __align__(16) unsigned char smem[SMEM_BYTES];
  unsigned short* actS = (unsigned short*)smem;                 // phase 1
  unsigned int*   actW32 = (unsigned int*)smem;
  unsigned short* hS   = (unsigned short*)smem;                 // phase 2
  unsigned short* wS   = (unsigned short*)(smem + WOFF_BYTES);  // both phases
  unsigned int*   wW32 = (unsigned int*)(smem + WOFF_BYTES);

  const int i = i0 + (int)blockIdx.x;
  const int j = tick - i;
  const int cell = i * GDIM + j;

  const int tid  = (int)threadIdx.x;
  const int wave = tid >> 5;
  const int lane = tid & 31;
  const int m    = lane & 15;       // A-row / B-col / C-col within tile
  const int half = (lane >> 4) & 1; // K-subset selector (ISA 16-bit layouts)
  const int koff = 8 * half;        // fragment K base for this lane half

  // Dependency sources: boundary x slots or previously computed grid cells.
  const float* leftBase; int leftStride;
  if (j == 0) { leftBase = x + i * DDIM;                            leftStride = 2 * GDIM * DDIM; }
  else        { leftBase = out + (size_t)(cell - 1) * BDIM * DDIM;  leftStride = DDIM; }
  const float* upBase; int upStride;
  if (i == 0) { upBase = x + (GDIM + j) * DDIM;                     upStride = 2 * GDIM * DDIM; }
  else        { upBase = out + (size_t)(cell - GDIM) * BDIM * DDIM; upStride = DDIM; }

  // Accumulators: wave's 16 rows x 192 cols = 12 C tiles; init with bias
  // (C element r maps to M = r + 8*half, N = m -> bias depends on N only).
  v8f acc[12];
#pragma unroll
  for (int nt = 0; nt < 12; ++nt) {
    float bv = bvec[cell * HDIM + nt * 16 + m];
#pragma unroll
    for (int r = 0; r < 8; ++r) acc[nt][r] = bv;
  }

  // ---------------- GEMM1: h = left@Wl + up@Wu + b ----------------
  for (int pass = 0; pass < 2; ++pass) {
    const float* aBase = pass ? upBase : leftBase;
    const int aStride  = pass ? upStride : leftStride;
    const float* wBase = (pass ? Wu : Wl) + (size_t)cell * DDIM * HDIM;
    for (int kk = 0; kk < 6; ++kk) {
      const int kbase = kk * 32;
      __syncthreads();  // previous chunk's LDS reads complete
      // Stage activation chunk [128][32] -> bf16 pairs (float2 loads, b32 stores).
#pragma unroll
      for (int e = 0; e < 8; ++e) {
        int idx = e * 256 + tid;         // pair index over 128x16
        int mr = idx >> 4, kp = idx & 15;
        fvec2 v2 = *(const fvec2*)(aBase + mr * aStride + kbase + kp * 2);
        actW32[mr * (CHUNK_STRIDE / 2) + kp] = packbf(v2.x, v2.y);
      }
      // Stage weight chunk transposed: wS[n][k] = W[kbase+k][n], packed K pairs.
#pragma unroll
      for (int e = 0; e < 12; ++e) {
        int idx = e * 256 + tid;         // pair index over 16x192
        int k2 = idx / HDIM, n = idx - k2 * HDIM;
        const float* wp = wBase + (kbase + k2 * 2) * HDIM + n;
        wW32[n * (CHUNK_STRIDE / 2) + k2] = packbf(wp[0], wp[HDIM]);
      }
      // Prefetch next weight chunk while WMMAs run (global_prefetch_b8).
      if (kk < 5) __builtin_prefetch(&wBase[(kbase + 32) * HDIM + tid * 24], 0, 0);
      __syncthreads();

      // A fragment: 2x ds_load_b128 (K 0..7 and 16..23, +8*half), bank-clean.
      Frag a;
      {
        const unsigned short* ab = actS + (wave * 16 + m) * CHUNK_STRIDE + koff;
        a.q[0] = *(const uivec4*)(ab);
        a.q[1] = *(const uivec4*)(ab + 16);
      }
      // Double-buffered B fragments: loads for nt+1 in flight across wmma(nt).
      Frag bfr[2];
      load_bfrag(bfr[0], wS, 0, m, koff);
#pragma unroll
      for (int nt = 0; nt < 12; ++nt) {
        if (nt < 11) load_bfrag(bfr[(nt + 1) & 1], wS, nt + 1, m, koff);
        acc[nt] = __builtin_amdgcn_wmma_f32_16x16x32_bf16(
            false, a.v, false, bfr[nt & 1].v, (short)0, acc[nt], false, false);
      }
    }
  }

  __syncthreads();  // GEMM1 LDS reads done; act region now reusable as h
  // Epilogue: exact gelu, pack bf16 h into LDS (A-major for GEMM2), zero acc.
#pragma unroll
  for (int nt = 0; nt < 12; ++nt) {
#pragma unroll
    for (int r = 0; r < 8; ++r) {
      int row = wave * 16 + 8 * half + r;
      hS[row * HBUF_STRIDE + nt * 16 + m] = f2bf(gelu_exact(acc[nt][r]));
    }
#pragma unroll
    for (int r = 0; r < 8; ++r) acc[nt][r] = 0.0f;
  }

  // ---------------- GEMM2: out = h @ V ----------------
  const float* vBase = V + (size_t)cell * HDIM * DDIM;
  for (int kk = 0; kk < 6; ++kk) {
    const int kbase = kk * 32;
    __syncthreads();
#pragma unroll
    for (int e = 0; e < 12; ++e) {
      int idx = e * 256 + tid;
      int k2 = idx / DDIM, n = idx - k2 * DDIM;
      const float* vp = vBase + (kbase + k2 * 2) * DDIM + n;
      wW32[n * (CHUNK_STRIDE / 2) + k2] = packbf(vp[0], vp[DDIM]);
    }
    if (kk < 5) __builtin_prefetch(&vBase[(kbase + 32) * DDIM + tid * 24], 0, 0);
    __syncthreads();

    // A fragment from h: 4x ds_load_b64 (stride 196 is 8B-aligned, bank-clean).
    Frag a;
    {
      const unsigned short* hb = hS + (wave * 16 + m) * HBUF_STRIDE + kbase + koff;
      a.d[0] = *(const uivec2*)(hb);
      a.d[1] = *(const uivec2*)(hb + 4);
      a.d[2] = *(const uivec2*)(hb + 16);
      a.d[3] = *(const uivec2*)(hb + 20);
    }
    Frag bfr[2];
    load_bfrag(bfr[0], wS, 0, m, koff);
#pragma unroll
    for (int nt = 0; nt < 12; ++nt) {
      if (nt < 11) load_bfrag(bfr[(nt + 1) & 1], wS, nt + 1, m, koff);
      acc[nt] = __builtin_amdgcn_wmma_f32_16x16x32_bf16(
          false, a.v, false, bfr[nt & 1].v, (short)0, acc[nt], false, false);
    }
  }

  // Store fp32 result grid[cell][b][d].
  float* outCell = out + (size_t)cell * BDIM * DDIM;
#pragma unroll
  for (int nt = 0; nt < 12; ++nt) {
#pragma unroll
    for (int r = 0; r < 8; ++r) {
      int row = wave * 16 + 8 * half + r;
      outCell[row * DDIM + nt * 16 + m] = acc[nt][r];
    }
  }
}

extern "C" void kernel_launch(void* const* d_in, const int* in_sizes, int n_in,
                              void* d_out, int out_size, void* d_ws, size_t ws_size,
                              hipStream_t stream) {
  (void)in_sizes; (void)n_in; (void)out_size; (void)d_ws; (void)ws_size;
  const float* x  = (const float*)d_in[0];
  const float* Wl = (const float*)d_in[1];
  const float* Wu = (const float*)d_in[2];
  const float* b  = (const float*)d_in[3];
  const float* V  = (const float*)d_in[4];
  float* out = (float*)d_out;

  // Anti-diagonal wavefront: stream-ordered launches provide the dependency.
  for (int t = 0; t < 2 * GDIM - 1; ++t) {
    int ilo = (t - (GDIM - 1)) > 0 ? (t - (GDIM - 1)) : 0;
    int ihi = t < (GDIM - 1) ? t : (GDIM - 1);
    int nc = ihi - ilo + 1;
    wavefront_cell_kernel<<<nc, 256, 0, stream>>>(x, Wl, Wu, b, V, out, t, ilo);
  }
}
